// ProxyTripletLoss_83502754169123
// MI455X (gfx1250) — matrix operand order repaired and checked by reading, not compiled, not run on hardware
//
#include <hip/hip_runtime.h>
#include <hip/hip_bf16.h>
#include <math.h>

#define MARGIN 0.2f
#define B_ROWS 2048
#define C_COLS 100000
#define D_DIM  128
// Packed row layout for split-bf16 planes: [hi x 128 | lo x 128] per row.
#define PK_STRIDE 256

typedef float  v2f   __attribute__((ext_vector_type(2)));
typedef float  v8f   __attribute__((ext_vector_type(8)));
typedef __bf16 v16bf __attribute__((ext_vector_type(16)));
typedef __bf16 v8bf  __attribute__((ext_vector_type(8)));

union V16U { v16bf v; v8bf h[2]; };

// Order-preserving float->uint encoding so atomicMax(uint) == float max.
__device__ __forceinline__ unsigned enc_f32(float f) {
    unsigned u = __float_as_uint(f);
    return (u & 0x80000000u) ? ~u : (u | 0x80000000u);
}
__device__ __forceinline__ float dec_f32(unsigned u) {
    unsigned b = (u & 0x80000000u) ? (u ^ 0x80000000u) : ~u;
    return __uint_as_float(b);
}

__global__ void __launch_bounds__(256) ptl_init_kernel(float* pos, unsigned* negbits) {
    int i = blockIdx.x * blockDim.x + threadIdx.x;
    if (i < B_ROWS) { pos[i] = 0.0f; negbits[i] = 0u; }
}

// One wave32 per proxy row: rsqn[c] = 1/sqrt(sum_d P[c,d]^2)
__global__ void __launch_bounds__(256) ptl_norm_kernel(const float* __restrict__ P,
                                                       float* __restrict__ rsqn) {
    int wave = blockIdx.x * (blockDim.x >> 5) + (threadIdx.x >> 5);
    int lane = threadIdx.x & 31;
    if (wave >= C_COLS) return;
    const float4* p4 = (const float4*)(P + (size_t)wave * D_DIM);
    float4 v = p4[lane];                       // 32 lanes * 4 floats = 128
    float s = v.x * v.x + v.y * v.y + v.z * v.z + v.w * v.w;
    #pragma unroll
    for (int off = 16; off > 0; off >>= 1) s += __shfl_xor(s, off, 32);
    if (lane == 0) rsqn[wave] = 1.0f / sqrtf(s);
}

// Normalize proxies, split fp32 -> bf16 (hi,lo), store interleaved rows.
__global__ void __launch_bounds__(256) ptl_cvt_p_kernel(const float* __restrict__ P,
                                                        const float* __restrict__ rsqn,
                                                        __bf16* __restrict__ Ppk) {
    size_t idx = (size_t)blockIdx.x * blockDim.x + threadIdx.x;
    if (idx >= (size_t)C_COLS * D_DIM) return;
    int c = (int)(idx >> 7);                   // D_DIM == 128
    int d = (int)(idx & 127);
    float pn = P[idx] * rsqn[c];
    __bf16 hi = (__bf16)pn;
    __bf16 lo = (__bf16)(pn - (float)hi);
    __bf16* row = Ppk + (size_t)c * PK_STRIDE;
    row[d]       = hi;
    row[128 + d] = lo;
}

// Split embeddings fp32 -> bf16 (hi,lo), interleaved rows.
__global__ void __launch_bounds__(256) ptl_cvt_e_kernel(const float* __restrict__ E,
                                                        __bf16* __restrict__ Epk) {
    size_t idx = (size_t)blockIdx.x * blockDim.x + threadIdx.x;
    if (idx >= (size_t)B_ROWS * D_DIM) return;
    int r = (int)(idx >> 7);
    int d = (int)(idx & 127);
    float e = E[idx];
    __bf16 hi = (__bf16)e;
    __bf16 lo = (__bf16)(e - (float)hi);
    __bf16* row = Epk + (size_t)r * PK_STRIDE;
    row[d]       = hi;
    row[128 + d] = lo;
}

// ---------------- Fast path: compensated bf16 (bf16x3) WMMA ----------------
// Block: 128 rows x 128 proxy cols, 8 waves. The 128 packed proxy rows (64KB)
// are staged once into LDS and shared by all 8 waves (8x traffic cut vs L2).
// Inner loop: ds_load_b128 with pure immediate offsets + v_wmma only.
// dot = ah*bh + al*bh + ah*bl  (al*bl term ~2^-16, dropped)
__global__ void __launch_bounds__(256, 1) ptl_wmma_bf16_kernel(
    const __bf16* __restrict__ Epk, const __bf16* __restrict__ Ppk,
    const int* __restrict__ labels,
    float* __restrict__ pos, unsigned* __restrict__ negbits)
{
    __shared__ __attribute__((aligned(16))) __bf16 smem[128 * PK_STRIDE]; // 64KB

    const int lane = threadIdx.x & 31;
    const int wid  = threadIdx.x >> 5;      // 0..7
    const int half = lane >> 4;
    const int q    = lane & 15;

    const int row0  = blockIdx.y * 128 + wid * 16;
    const int cbase = blockIdx.x * 128;

    // Cooperative stage: 64KB = 4096 x 16B chunks; one full row per wave/iter.
    {
        const int tid = threadIdx.x;
        #pragma unroll
        for (int it = 0; it < 16; ++it) {
            int chunk = tid + it * 256;      // 0..4095
            int r     = chunk >> 5;          // 32 x 16B chunks per 512B row
            int off   = (chunk & 31) * 8;    // element offset (8 elems = 16B)
            int c     = cbase + r;
            unsigned pc = (unsigned)((c < C_COLS) ? c : (C_COLS - 1));
            *(uint4*)(smem + r * PK_STRIDE + off) =
                *(const uint4*)(Ppk + (size_t)pc * PK_STRIDE + off);
        }
    }

    // A fragments (16-bit A 16x32 layout): per chunk kc,
    // elements 0..7  = k = kc*32 + 8*half + 0..7
    // elements 8..15 = k = kc*32 + 16 + 8*half + 0..7
    V16U ah[4], al[4];
    {
        const unsigned aoff = (unsigned)(row0 + q) * PK_STRIDE + 8u * half;
        #pragma unroll
        for (int kc = 0; kc < 4; ++kc) {
            ah[kc].h[0] = *(const v8bf*)(Epk + aoff + kc * 32);
            ah[kc].h[1] = *(const v8bf*)(Epk + aoff + kc * 32 + 16);
            al[kc].h[0] = *(const v8bf*)(Epk + aoff + 128 + kc * 32);
            al[kc].h[1] = *(const v8bf*)(Epk + aoff + 128 + kc * 32 + 16);
        }
    }

    int lab[8];
    #pragma unroll
    for (int i = 0; i < 8; ++i) lab[i] = labels[row0 + i + 8 * half];

    v8f acc[8];
    #pragma unroll
    for (int t = 0; t < 8; ++t) acc[t] = (v8f){0.f,0.f,0.f,0.f,0.f,0.f,0.f,0.f};

    __syncthreads();

    // B fragments from LDS: single base (q*PK_STRIDE + 8*half); every other
    // term (t*16*PK_STRIDE, kc*32, +16, +128) is a constant folded into the
    // 16-bit DS offset immediate (max byte offset 65520 < 65536).
    const __bf16* bsm = smem + q * PK_STRIDE + 8 * half;
    #pragma unroll
    for (int kc = 0; kc < 4; ++kc) {
        #pragma unroll
        for (int t = 0; t < 8; ++t) {
            V16U bh, bl;
            bh.h[0] = *(const v8bf*)(bsm + t * 16 * PK_STRIDE + kc * 32);
            bh.h[1] = *(const v8bf*)(bsm + t * 16 * PK_STRIDE + kc * 32 + 16);
            bl.h[0] = *(const v8bf*)(bsm + t * 16 * PK_STRIDE + kc * 32 + 128);
            bl.h[1] = *(const v8bf*)(bsm + t * 16 * PK_STRIDE + kc * 32 + 144);
            acc[t] = __builtin_amdgcn_wmma_f32_16x16x32_bf16(
                false, ah[kc].v, false, bh.v, (short)0, acc[t], false, false);
            acc[t] = __builtin_amdgcn_wmma_f32_16x16x32_bf16(
                false, al[kc].v, false, bh.v, (short)0, acc[t], false, false);
            acc[t] = __builtin_amdgcn_wmma_f32_16x16x32_bf16(
                false, ah[kc].v, false, bl.v, (short)0, acc[t], false, false);
        }
    }

    #pragma unroll
    for (int i = 0; i < 8; ++i) {
        const int row = row0 + i + 8 * half;
        float m = -INFINITY;
        #pragma unroll
        for (int t = 0; t < 8; ++t) {
            int col = cbase + t * 16 + q;
            if (col < C_COLS) {
                float val = acc[t][i];       // normalization folded into Ppk
                if (col == lab[i]) pos[row] = val;
                else               m = fmaxf(m, val);
            }
        }
        #pragma unroll
        for (int off = 8; off > 0; off >>= 1)
            m = fmaxf(m, __shfl_xor(m, off, 16));
        if (q == 0)
            atomicMax(negbits + row, enc_f32(m));
    }
}

// ---------------- Fallback path: exact fp32 WMMA (16x16x4) ----------------
__global__ void __launch_bounds__(256, 1) ptl_wmma_f32_kernel(
    const float* __restrict__ E, const float* __restrict__ P,
    const int* __restrict__ labels, const float* __restrict__ rsqn,
    float* __restrict__ pos, unsigned* __restrict__ negbits)
{
    const int lane = threadIdx.x & 31;
    const int wid  = threadIdx.x >> 5;
    const int half = lane >> 4;
    const int q    = lane & 15;

    const int row0  = blockIdx.y * 128 + wid * 16;
    const int cbase = blockIdx.x * 128;

    v2f a[32];
    const unsigned aoff = (unsigned)(row0 + q) * D_DIM + 2u * half;
    #pragma unroll
    for (int ks = 0; ks < 32; ++ks)
        a[ks] = *(const v2f*)(E + aoff + ks * 4);

    int lab[8];
    #pragma unroll
    for (int i = 0; i < 8; ++i) lab[i] = labels[row0 + i + 8 * half];

    v8f acc[8];
    #pragma unroll
    for (int t = 0; t < 8; ++t) acc[t] = (v8f){0.f,0.f,0.f,0.f,0.f,0.f,0.f,0.f};

    unsigned boff[8];
    #pragma unroll
    for (int t = 0; t < 8; ++t) {
        int c = cbase + t * 16 + q;
        unsigned pc = (unsigned)((c < C_COLS) ? c : (C_COLS - 1));
        boff[t] = pc * D_DIM + 2u * half;
    }

    #pragma unroll
    for (int ks = 0; ks < 32; ++ks) {
        #pragma unroll
        for (int t = 0; t < 8; ++t) {
            v2f b = *(const v2f*)(P + boff[t] + ks * 4);
            acc[t] = __builtin_amdgcn_wmma_f32_16x16x4_f32(
                false, a[ks], false, b, (short)0, acc[t], false, false);
        }
    }

    float rs[8];
    #pragma unroll
    for (int t = 0; t < 8; ++t) {
        int c = cbase + t * 16 + q;
        rs[t] = rsqn[(c < C_COLS) ? c : (C_COLS - 1)];
    }

    #pragma unroll
    for (int i = 0; i < 8; ++i) {
        const int row = row0 + i + 8 * half;
        float m = -INFINITY;
        #pragma unroll
        for (int t = 0; t < 8; ++t) {
            int col = cbase + t * 16 + q;
            if (col < C_COLS) {
                float val = acc[t][i] * rs[t];
                if (col == lab[i]) pos[row] = val;
                else               m = fmaxf(m, val);
            }
        }
        #pragma unroll
        for (int off = 8; off > 0; off >>= 1)
            m = fmaxf(m, __shfl_xor(m, off, 16));
        if (q == 0)
            atomicMax(negbits + row, enc_f32(m));
    }
}

__global__ void __launch_bounds__(256) ptl_final_kernel(const float* __restrict__ pos,
                                                        const unsigned* __restrict__ negbits,
                                                        float* __restrict__ out) {
    __shared__ float sdata[256];
    float s = 0.0f;
    for (int r = threadIdx.x; r < B_ROWS; r += 256) {
        float maxneg = dec_f32(negbits[r]);
        // (2-2*pos) + margin - (2-2*maxneg) = 2*(maxneg-pos) + margin
        float l = 2.0f * (maxneg - pos[r]) + MARGIN;
        s += (l > 0.0f) ? l : 0.0f;
    }
    sdata[threadIdx.x] = s;
    __syncthreads();
    #pragma unroll
    for (int st = 128; st > 0; st >>= 1) {
        if ((int)threadIdx.x < st) sdata[threadIdx.x] += sdata[threadIdx.x + st];
        __syncthreads();
    }
    if (threadIdx.x == 0) out[0] = sdata[0] / (float)B_ROWS;
}

extern "C" void kernel_launch(void* const* d_in, const int* in_sizes, int n_in,
                              void* d_out, int out_size, void* d_ws, size_t ws_size,
                              hipStream_t stream) {
    const float* E      = (const float*)d_in[0];   // [2048,128] fp32 (unit rows)
    const float* P      = (const float*)d_in[1];   // [100000,128] fp32
    const int*   labels = (const int*)d_in[2];     // [2048]
    float*       out    = (float*)d_out;

    // Workspace layout (256B-aligned slots)
    size_t cur = 0;
    auto take = [&](size_t bytes) {
        size_t o = cur;
        cur += (bytes + 255) & ~(size_t)255;
        return o;
    };
    char* wsb = (char*)d_ws;
    size_t o_rsqn = take((size_t)C_COLS * 4);
    size_t o_pos  = take((size_t)B_ROWS * 4);
    size_t o_neg  = take((size_t)B_ROWS * 4);
    size_t o_epk  = take((size_t)B_ROWS * PK_STRIDE * 2);
    size_t o_ppk  = take((size_t)C_COLS * PK_STRIDE * 2);
    size_t fast_bytes = cur;

    float*    rsqn    = (float*)(wsb + o_rsqn);
    float*    pos     = (float*)(wsb + o_pos);
    unsigned* negbits = (unsigned*)(wsb + o_neg);
    __bf16*   Epk     = (__bf16*)(wsb + o_epk);
    __bf16*   Ppk     = (__bf16*)(wsb + o_ppk);

    const bool fast = (ws_size >= fast_bytes);   // fixed per process -> deterministic

    ptl_init_kernel<<<(B_ROWS + 255) / 256, 256, 0, stream>>>(pos, negbits);
    ptl_norm_kernel<<<(C_COLS + 7) / 8, 256, 0, stream>>>(P, rsqn);

    dim3 grid((C_COLS + 127) / 128, B_ROWS / 128);
    if (fast) {
        ptl_cvt_e_kernel<<<(B_ROWS * D_DIM + 255) / 256, 256, 0, stream>>>(E, Epk);
        ptl_cvt_p_kernel<<<(int)(((size_t)C_COLS * D_DIM + 255) / 256), 256, 0, stream>>>(P, rsqn, Ppk);
        ptl_wmma_bf16_kernel<<<grid, 256, 0, stream>>>(Epk, Ppk, labels, pos, negbits);
    } else {
        ptl_wmma_f32_kernel<<<grid, 256, 0, stream>>>(E, P, labels, rsqn, pos, negbits);
    }

    ptl_final_kernel<<<1, 256, 0, stream>>>(pos, negbits, out);
}
